// GCNLayer_69552700391515
// MI455X (gfx1250) — compile-verified, matrix operand order checked by reading, not run on hardware
//
#include <hip/hip_runtime.h>
#include <hip/hip_bf16.h>

#define DIM 128
#define BN_EPS 1e-5f

typedef __attribute__((ext_vector_type(16))) __bf16 v16bf;
typedef __attribute__((ext_vector_type(8)))  float  v8f;

// ---------------------------------------------------------------------------
// Kernel 0: zero agg accumulator + stats (ws is poisoned by harness)
// ---------------------------------------------------------------------------
__global__ void gcn_zero_kernel(float* p, int total) {
    int stride = gridDim.x * blockDim.x;
    for (int i = blockIdx.x * blockDim.x + threadIdx.x; i < total; i += stride)
        p[i] = 0.0f;
}

// ---------------------------------------------------------------------------
// Kernel 1: gather features[src] rows, scatter-add into agg[dst] rows.
// One wave32 per edge: lane l moves float4 at column 4*l (32*4 = 128 = DIM).
// Coalesced 512B row reads; f32 global atomics for the segment sum.
// ---------------------------------------------------------------------------
__global__ void gcn_scatter_kernel(const float* __restrict__ feat,
                                   const int* __restrict__ src,
                                   const int* __restrict__ dst,
                                   float* __restrict__ agg, int E) {
    int wave   = (blockIdx.x * blockDim.x + threadIdx.x) >> 5;
    int lane   = threadIdx.x & 31;
    int nWaves = (gridDim.x * blockDim.x) >> 5;
    for (int e = wave; e < E; e += nWaves) {
        int s = src[e];
        int d = dst[e];
        if (e + nWaves < E) {
            // hint the next gather row toward L2 (global_prefetch_b8)
            __builtin_prefetch(&feat[(size_t)src[e + nWaves] * DIM], 0, 1);
        }
        float4 v = ((const float4*)(feat + (size_t)s * DIM))[lane];
        float* o = agg + (size_t)d * DIM + lane * 4;
        atomicAdd(o + 0, v.x);
        atomicAdd(o + 1, v.y);
        atomicAdd(o + 2, v.z);
        atomicAdd(o + 3, v.w);
    }
}

__device__ __forceinline__ void cvt8(v16bf& f, int base, float4 a, float4 b) {
    f[base + 0] = (__bf16)a.x; f[base + 1] = (__bf16)a.y;
    f[base + 2] = (__bf16)a.z; f[base + 3] = (__bf16)a.w;
    f[base + 4] = (__bf16)b.x; f[base + 5] = (__bf16)b.y;
    f[base + 6] = (__bf16)b.z; f[base + 7] = (__bf16)b.w;
}

// ---------------------------------------------------------------------------
// Kernel 2: h = agg @ W via v_wmma_f32_16x16x32_bf16, fused BN column stats.
// Block = 256 threads = 8 waves; block covers 128 rows x 128 cols.
// Each wave owns 16 rows, sweeps 8 column tiles, K = 128 in 4 steps of 32.
//
// W is staged into LDS already packed in WMMA B-fragment order, so the inner
// loop fetches each B fragment with two ds_load_b128 (one v16bf deref).
// Per-block BN column sum/sumsq reduced in LDS (ds_add_f32), one global
// atomic per column per block.
// ---------------------------------------------------------------------------
__global__ void gcn_gemm_kernel(const float* __restrict__ agg,
                                const float* __restrict__ Wg,
                                float* __restrict__ h,
                                float* __restrict__ stats, int N) {
    // Wp[((tile*4 + ks)*32 + lane)*16 + e] : e = 16 bf16 of lane's B fragment
    __shared__ __bf16 Wp[8 * 4 * 32 * 16];   // 32 KB
    __shared__ float  sstat[2 * DIM];        // [0,128): sum, [128,256): sumsq

    const int tid  = threadIdx.x;
    const int lane = tid & 31;
    const int wave = tid >> 5;
    const int koff = (lane & 16) ? 8 : 0;    // ISA 16-bit K split between halves

    // ---- stage W -> LDS in B-fragment layout; wave w packs column-tile w ---
    {
        const int c = wave * 16 + (lane & 15);       // B layout: lane -> column
#pragma unroll
        for (int ks = 0; ks < 4; ++ks) {
            const int kb = ks * 32 + koff;
            __bf16* dp = Wp + ((wave * 4 + ks) * 32 + lane) * 16;
#pragma unroll
            for (int i = 0; i < 8; ++i) {            // coalesced 64B row reads
                dp[i]     = (__bf16)Wg[(kb + i)      * DIM + c];
                dp[8 + i] = (__bf16)Wg[(kb + 16 + i) * DIM + c];
            }
        }
    }
    if (tid < 2 * DIM) sstat[tid] = 0.0f;
    __syncthreads();

    const int row0 = blockIdx.x * 128 + wave * 16;
    const int r    = row0 + (lane & 15);
    // wave-uniform conditions, scalarized so no EXEC masking in hot paths
    const bool fullTile = __builtin_amdgcn_readfirstlane((int)(row0 + 16 <= N));
    const bool anyRow   = __builtin_amdgcn_readfirstlane((int)(row0 < N));

    // ---- A fragments for all 4 K-steps (16-bit A 16x32 ISA layout) --------
    v16bf afrag[4] = {};
    if (fullTile) {
        const float4* a4 = (const float4*)(agg + (size_t)r * DIM);
#pragma unroll
        for (int ks = 0; ks < 4; ++ks) {
            const int kb = ks * 32 + koff;           // multiple of 4
            cvt8(afrag[ks], 0, a4[kb >> 2],        a4[(kb >> 2) + 1]);
            cvt8(afrag[ks], 8, a4[(kb + 16) >> 2], a4[((kb + 16) >> 2) + 1]);
        }
    } else if (anyRow) {
        const bool rv = (r < N);
        const float* arow = agg + (size_t)r * DIM;
#pragma unroll
        for (int ks = 0; ks < 4; ++ks) {
            const int kb = ks * 32 + koff;
#pragma unroll
            for (int i = 0; i < 8; ++i) {
                afrag[ks][i]     = (__bf16)(rv ? arow[kb + i]      : 0.0f);
                afrag[ks][8 + i] = (__bf16)(rv ? arow[kb + 16 + i] : 0.0f);
            }
        }
    }

    const int mrow = row0 + ((lane & 16) ? 8 : 0);   // C/D layout M origin

#pragma unroll
    for (int t = 0; t < 8; ++t) {
        const int c = t * 16 + (lane & 15);
        v8f acc = {};
#pragma unroll
        for (int ks = 0; ks < 4; ++ks) {
            const v16bf bfrag =
                *(const v16bf*)(Wp + ((t * 4 + ks) * 32 + lane) * 16);
            // (neg_a, A, neg_b, B, c_mod, C, reuse_a, reuse_b)
            acc = __builtin_amdgcn_wmma_f32_16x16x32_bf16(
                false, afrag[ks], false, bfrag, (short)0, acc, false, false);
        }
        if (fullTile) {                               // fast path: no guards
            float s = 0.0f, sq = 0.0f;
#pragma unroll
            for (int v = 0; v < 8; ++v) {
                float x = acc[v];
                h[(size_t)(mrow + v) * DIM + c] = x;
                s  += x;
                sq += x * x;
            }
            atomicAdd(&sstat[c], s);                  // ds_add_f32
            atomicAdd(&sstat[DIM + c], sq);
        } else if (anyRow) {                          // rare tail path
            float s = 0.0f, sq = 0.0f;
#pragma unroll
            for (int v = 0; v < 8; ++v) {
                const int rr = mrow + v;
                if (rr < N) {
                    float x = acc[v];
                    h[(size_t)rr * DIM + c] = x;
                    s  += x;
                    sq += x * x;
                }
            }
            atomicAdd(&sstat[c], s);
            atomicAdd(&sstat[DIM + c], sq);
        }
    }

    __syncthreads();
    if (tid < 2 * DIM) atomicAdd(&stats[tid], sstat[tid]);
}

// ---------------------------------------------------------------------------
// Kernel 3: out = feat + relu((h - mean) * rsqrt(var+eps) * gamma + beta)
// (linear bias b cancels exactly inside BatchNorm -> dropped)
// float4 per thread: 4 consecutive columns of one row.
// ---------------------------------------------------------------------------
__global__ void gcn_finalize_kernel(const float* __restrict__ h,
                                    const float* __restrict__ feat,
                                    const float* __restrict__ gamma,
                                    const float* __restrict__ beta,
                                    const float* __restrict__ stats,
                                    float* __restrict__ out, int N) {
    int q      = blockIdx.x * blockDim.x + threadIdx.x;   // quad index
    int totalQ = N * (DIM / 4);
    if (q >= totalQ) return;
    int    col0 = (q * 4) & (DIM - 1);
    float  invN = 1.0f / (float)N;
    float4 hv   = ((const float4*)h)[q];
    float4 fv   = ((const float4*)feat)[q];
    float  r[4] = {hv.x, hv.y, hv.z, hv.w};
    float  f[4] = {fv.x, fv.y, fv.z, fv.w};
#pragma unroll
    for (int i = 0; i < 4; ++i) {
        int   c    = col0 + i;
        float mean = stats[c] * invN;
        float var  = stats[DIM + c] * invN - mean * mean;
        float inv  = rsqrtf(var + BN_EPS);
        float v    = (r[i] - mean) * inv * gamma[c] + beta[c];
        r[i]       = f[i] + fmaxf(v, 0.0f);
    }
    float4 o = {r[0], r[1], r[2], r[3]};
    ((float4*)out)[q] = o;
}

// ---------------------------------------------------------------------------
extern "C" void kernel_launch(void* const* d_in, const int* in_sizes, int n_in,
                              void* d_out, int out_size, void* d_ws, size_t ws_size,
                              hipStream_t stream) {
    const float* feat  = (const float*)d_in[0];
    const float* W     = (const float*)d_in[1];
    // d_in[2] = b : unused, cancels in batch-norm
    const float* gamma = (const float*)d_in[3];
    const float* beta  = (const float*)d_in[4];
    const int*   src   = (const int*)d_in[5];
    const int*   dst   = (const int*)d_in[6];
    float*       out   = (float*)d_out;

    const int N = in_sizes[0] / DIM;  // 20000
    const int E = in_sizes[5];        // 640000

    // workspace layout: [agg N*DIM | stats 256 | h N*DIM]  (~20.5 MB)
    float* ws    = (float*)d_ws;
    float* agg   = ws;
    float* stats = ws + (size_t)N * DIM;
    float* h     = stats + 2 * DIM;

    const int zTotal  = N * DIM + 2 * DIM;     // zero agg + stats in one pass
    const int zBlocks = (zTotal + 255) / 256;
    gcn_zero_kernel<<<zBlocks, 256, 0, stream>>>(ws, zTotal);

    gcn_scatter_kernel<<<2048, 256, 0, stream>>>(feat, src, dst, agg, E);

    const int gBlocks = (N + 127) / 128;       // 157
    gcn_gemm_kernel<<<gBlocks, 256, 0, stream>>>(agg, W, h, stats, N);

    const int fBlocks = (N * (DIM / 4) + 255) / 256;
    gcn_finalize_kernel<<<fBlocks, 256, 0, stream>>>(h, feat, gamma, beta,
                                                     stats, out, N);
}
